// CustomGRU_76063870812449
// MI455X (gfx1250) — matrix operand order repaired and checked by reading, not compiled, not run on hardware
//
#include <hip/hip_runtime.h>
#include <hip/hip_bf16.h>
#include <math.h>

// ---------------------------------------------------------------------------
// CDNA5 (gfx1250, wave32) bf16 WMMA GRU
//   B=64, S=512, I=H=1024, L=2, gates ordered r,z,n
// ---------------------------------------------------------------------------

typedef __attribute__((ext_vector_type(16))) __bf16   v16bf;
typedef __attribute__((ext_vector_type(8)))  float    v8f;
typedef __attribute__((ext_vector_type(4)))  unsigned u32x4;

union Frag { v16bf v; u32x4 q[2]; };

constexpr int Bn = 64, Sn = 512, In = 1024, Hn = 1024, G3 = 3072;
constexpr int Mrows = Bn * Sn;           // 32768 rows for the input-projection GEMM

static __device__ __forceinline__ float sigmoidf_(float x) {
    return 1.0f / (1.0f + __expf(-x));
}

// A-fragment (16x32 bf16, MxK): lane = M row; lane-half selects K sub-ranges
static __device__ __forceinline__ void loadA(Frag& f, const __bf16* __restrict__ row,
                                             int kb, int half) {
    f.q[0] = *(const u32x4*)(row + kb + half * 8);        // K = kb+half*8   .. +7
    f.q[1] = *(const u32x4*)(row + kb + 16 + half * 8);   // K = kb+16+half*8.. +7
}
// B-fragment (32x16 bf16, KxN): lane = N col; lane-half selects K sub-ranges (same as A)
static __device__ __forceinline__ void loadB(Frag& f, const __bf16* __restrict__ row,
                                             int kb, int half) {
    const __bf16* p = row + kb + half * 8;                // K = kb+half*8 .. +7, kb+16+half*8 .. +7
    f.q[0] = *(const u32x4*)(p);
    f.q[1] = *(const u32x4*)(p + 16);
}
static __device__ __forceinline__ v8f wmma_bf16(const Frag& a, const Frag& b, v8f c) {
    return __builtin_amdgcn_wmma_f32_16x16x32_bf16(false, a.v, false, b.v,
                                                   (short)0, c, false, false);
}

// ---------------------------------------------------------------------------
// f32 -> bf16 conversion (weights once, X once)
// ---------------------------------------------------------------------------
__global__ void cvt_f32_bf16(const float* __restrict__ src, __bf16* __restrict__ dst, int n) {
    int i = blockIdx.x * blockDim.x + threadIdx.x;
    const int stride = gridDim.x * blockDim.x;
    for (; i < n; i += stride) dst[i] = (__bf16)src[i];
}

// ---------------------------------------------------------------------------
// out[m,g] = sum_k A[m,k] * W[g,k] + bias[g]
// A: [M,K] bf16 row-major, W: [N,K] bf16 row-major (PyTorch W_ih layout).
// One wave computes a 32x64 output strip (2 A-frags x 4 B-frags = 8 WMMAs per
// K-step), with double-buffered fragments so next K-step loads overlap WMMAs.
// ---------------------------------------------------------------------------
__global__ void __launch_bounds__(256) gemm_bias_bf16(
    const __bf16* __restrict__ A, const __bf16* __restrict__ W,
    const float* __restrict__ bias, float* __restrict__ out,
    int M, int N, int K)
{
    const int lane    = threadIdx.x & 31;
    const int waveId  = blockIdx.x * 8 + (threadIdx.x >> 5);
    const int nGroups = N >> 6;                 // 64 cols per wave
    const int mTile   = waveId / nGroups;       // 32 rows per wave
    const int nGroup  = waveId % nGroups;
    if (mTile * 32 >= M) return;
    const int rA   = lane & 15;
    const int half = lane >> 4;

    const __bf16* arow0 = A + (size_t)(mTile * 32 + rA) * K;
    const __bf16* arow1 = arow0 + (size_t)16 * K;
    const __bf16* wrow  = W + (size_t)(nGroup * 64 + rA) * K;

    v8f acc[2][4] = {{{}, {}, {}, {}}, {{}, {}, {}, {}}};
    Frag a_c[2], b_c[4], a_n[2], b_n[4];

    loadA(a_c[0], arow0, 0, half);
    loadA(a_c[1], arow1, 0, half);
#pragma unroll
    for (int s = 0; s < 4; ++s) loadB(b_c[s], wrow + (size_t)s * 16 * K, 0, half);

    for (int kb = 0; kb < K - 32; kb += 32) {
        // prefetch next K-step into the shadow register set
        loadA(a_n[0], arow0, kb + 32, half);
        loadA(a_n[1], arow1, kb + 32, half);
#pragma unroll
        for (int s = 0; s < 4; ++s) loadB(b_n[s], wrow + (size_t)s * 16 * K, kb + 32, half);
        // 8 WMMAs on the current set (loads above remain in flight)
#pragma unroll
        for (int s = 0; s < 4; ++s) {
            acc[0][s] = wmma_bf16(a_c[0], b_c[s], acc[0][s]);
            acc[1][s] = wmma_bf16(a_c[1], b_c[s], acc[1][s]);
        }
        a_c[0] = a_n[0]; a_c[1] = a_n[1];
#pragma unroll
        for (int s = 0; s < 4; ++s) b_c[s] = b_n[s];
    }
#pragma unroll
    for (int s = 0; s < 4; ++s) {           // final K-step
        acc[0][s] = wmma_bf16(a_c[0], b_c[s], acc[0][s]);
        acc[1][s] = wmma_bf16(a_c[1], b_c[s], acc[1][s]);
    }

    const int colBase = nGroup * 64 + rA;
#pragma unroll
    for (int mt = 0; mt < 2; ++mt) {
#pragma unroll
        for (int s = 0; s < 4; ++s) {
            const int g  = colBase + s * 16;
            const float bv = bias[g];
#pragma unroll
            for (int r = 0; r < 8; ++r) {
                const int m = mTile * 32 + mt * 16 + half * 8 + r;
                out[(size_t)m * N + g] = acc[mt][s][r] + bv;
            }
        }
    }
}

// ---------------------------------------------------------------------------
// zero h state (f32 + both bf16 ping-pong buffers) and barrier counters
// ---------------------------------------------------------------------------
__global__ void zero_state(float* __restrict__ h, __bf16* __restrict__ hbf,
                           unsigned* __restrict__ bar) {
    const int i = blockIdx.x * blockDim.x + threadIdx.x;
    if (i < Bn * Hn)     h[i]   = 0.0f;
    if (i < 2 * Bn * Hn) hbf[i] = (__bf16)0.0f;
    if (i < 2)           bar[i] = 0u;
}

// ---------------------------------------------------------------------------
// Persistent recurrent kernel: 32 blocks x 8 waves = 256 tiles covering the
// 64x3072 per-step hidden GEMM. Each wave owns one 16-wide hidden column tile
// and computes all three gate tiles (r,z,n) so gate fusion is wave-local.
// h (bf16) rows for the block's M-tile are staged into LDS each step; bf16 h
// is double-buffered so a single grid barrier per step suffices. Fragment
// loads are double-buffered across K-steps to overlap with the WMMAs.
// ---------------------------------------------------------------------------
__global__ void __launch_bounds__(256) gru_recurrent(
    const float* __restrict__ xg,      // [B,S,3H] precomputed input projections
    const __bf16* __restrict__ Whh,    // [3H,H] bf16
    const float* __restrict__ bhh,     // [3H]
    float* __restrict__ h,             // [B,H] f32 state
    __bf16* __restrict__ hbf,          // [2][B,H] bf16 state (ping-pong)
    __bf16* y_bf,                      // [B,S,H] bf16 output (or null)
    float*  y_f,                       // [B,S,H] f32 output (or null)
    float* __restrict__ h_final,       // [B,H] final hidden -> d_out
    unsigned* __restrict__ bar,        // {count, generation}
    int S)
{
    __shared__ __bf16 hsh[16 * Hn];    // 32 KB: this block's 16 h rows in bf16

    const int lane  = threadIdx.x & 31;
    const int wave  = threadIdx.x >> 5;
    const int mBase = (blockIdx.x >> 3) * 16;                  // 4 M-tiles
    const int nBase = ((((blockIdx.x & 7) << 3) | wave)) * 16; // 64 N-tiles
    const int rA    = lane & 15;
    const int half  = lane >> 4;
    const int col   = nBase + rA;

    const __bf16* w_r = Whh + (size_t)(0 * Hn + nBase + rA) * Hn;
    const __bf16* w_z = Whh + (size_t)(1 * Hn + nBase + rA) * Hn;
    const __bf16* w_n = Whh + (size_t)(2 * Hn + nBase + rA) * Hn;
    const __bf16* ash = &hsh[rA * Hn];
    const float bhr = bhh[col], bhz = bhh[Hn + col], bhn = bhh[2 * Hn + col];

    for (int t = 0; t < S; ++t) {
        { // stage current bf16 h rows [mBase, mBase+16) -> LDS (linear 32KB copy)
            const u32x4* src = (const u32x4*)(hbf + (size_t)(t & 1) * (Bn * Hn)
                                                  + (size_t)mBase * Hn);
            u32x4* dst = (u32x4*)hsh;
            for (int i = threadIdx.x; i < (16 * Hn) / 8; i += 256) dst[i] = src[i];
        }
        __syncthreads();

        v8f ar = {}, az = {}, an = {};
        Frag a_c, br_c, bz_c, bn_c, a_n, br_n, bz_n, bn_n;
        loadA(a_c, ash, 0, half);
        loadB(br_c, w_r, 0, half);
        loadB(bz_c, w_z, 0, half);
        loadB(bn_c, w_n, 0, half);

        for (int kb = 0; kb < Hn - 32; kb += 32) {
            loadA(a_n, ash, kb + 32, half);          // prefetch next K-step
            loadB(br_n, w_r, kb + 32, half);
            loadB(bz_n, w_z, kb + 32, half);
            loadB(bn_n, w_n, kb + 32, half);
            ar = wmma_bf16(a_c, br_c, ar);
            az = wmma_bf16(a_c, bz_c, az);
            an = wmma_bf16(a_c, bn_c, an);
            a_c = a_n; br_c = br_n; bz_c = bz_n; bn_c = bn_n;
        }
        ar = wmma_bf16(a_c, br_c, ar);               // final K-step
        az = wmma_bf16(a_c, bz_c, az);
        an = wmma_bf16(a_c, bn_c, an);

        __bf16* hbf_next = hbf + (size_t)((t + 1) & 1) * (Bn * Hn);
#pragma unroll
        for (int r = 0; r < 8; ++r) {
            const int m = mBase + half * 8 + r;
            const size_t xoff = ((size_t)m * S + t) * (size_t)G3;
            const float xr = xg[xoff + col];
            const float xz = xg[xoff + Hn + col];
            const float xn = xg[xoff + 2 * Hn + col];
            const float rg = sigmoidf_(xr + ar[r] + bhr);
            const float zg = sigmoidf_(xz + az[r] + bhz);
            const float ng = tanhf(xn + rg * (an[r] + bhn));
            const float hprev = h[(size_t)m * Hn + col];
            const float hnew  = (1.0f - zg) * ng + zg * hprev;
            h[(size_t)m * Hn + col]        = hnew;
            hbf_next[(size_t)m * Hn + col] = (__bf16)hnew;
            const size_t yoff = ((size_t)m * S + t) * (size_t)Hn + col;
            if (y_bf) y_bf[yoff] = (__bf16)hnew;
            if (y_f)  y_f[yoff]  = hnew;
            if (t == S - 1) h_final[(size_t)m * Hn + col] = hnew;
        }

        // ---- grid-wide barrier (count + generation, sleep while spinning) ----
        __threadfence();
        __syncthreads();
        if (threadIdx.x == 0) {
            const unsigned g = ((volatile unsigned*)bar)[1];
            const unsigned prev = atomicAdd(&bar[0], 1u);
            if (prev + 1 == gridDim.x) {
                ((volatile unsigned*)bar)[0] = 0;
                __threadfence();
                atomicAdd(&bar[1], 1u);
            } else {
                while (((volatile unsigned*)bar)[1] == g) __builtin_amdgcn_s_sleep(1);
            }
            __threadfence();
        }
        __syncthreads();
    }
}

// ---------------------------------------------------------------------------
// Host-side orchestration (all on `stream`, graph-capture safe)
// ---------------------------------------------------------------------------
extern "C" void kernel_launch(void* const* d_in, const int* in_sizes, int n_in,
                              void* d_out, int out_size, void* d_ws, size_t ws_size,
                              hipStream_t stream)
{
    (void)in_sizes; (void)n_in; (void)out_size; (void)ws_size;

    const float* X    = (const float*)d_in[0];
    const float* Wih0 = (const float*)d_in[1];
    const float* bih0 = (const float*)d_in[2];
    const float* Whh0 = (const float*)d_in[3];
    const float* bhh0 = (const float*)d_in[4];
    const float* Wih1 = (const float*)d_in[5];
    const float* bih1 = (const float*)d_in[6];
    const float* Whh1 = (const float*)d_in[7];
    const float* bhh1 = (const float*)d_in[8];

    char* ws = (char*)d_ws;
    constexpr size_t XG_BYTES  = (size_t)Mrows * G3 * 4;   // 384 MB f32 preactivations
    constexpr size_t ACT_BYTES = (size_t)Mrows * Hn * 2;   // 64 MB bf16 activations
    constexpr size_t W_ELEMS   = (size_t)G3 * Hn;

    float*    xg  = (float*)(ws);
    __bf16*   act = (__bf16*)(ws + XG_BYTES);
    __bf16*   w0i = (__bf16*)(ws + XG_BYTES + ACT_BYTES);
    __bf16*   w0h = w0i + W_ELEMS;
    __bf16*   w1i = w0h + W_ELEMS;
    __bf16*   w1h = w1i + W_ELEMS;
    float*    h   = (float*)(w1h + W_ELEMS);
    __bf16*   hbf = (__bf16*)(h + Bn * Hn);
    unsigned* bar = (unsigned*)(hbf + 2 * Bn * Hn);

    float* y1     = (float*)d_out;                         // [B,S,H]
    float* h0_out = y1 + (size_t)Bn * Sn * Hn;             // [B,H]
    float* h1_out = h0_out + (size_t)Bn * Hn;              // [B,H]

    // one-time conversions to bf16
    cvt_f32_bf16<<<2048, 256, 0, stream>>>(X, act, Mrows * In);
    cvt_f32_bf16<<<512, 256, 0, stream>>>(Wih0, w0i, (int)W_ELEMS);
    cvt_f32_bf16<<<512, 256, 0, stream>>>(Whh0, w0h, (int)W_ELEMS);
    cvt_f32_bf16<<<512, 256, 0, stream>>>(Wih1, w1i, (int)W_ELEMS);
    cvt_f32_bf16<<<512, 256, 0, stream>>>(Whh1, w1h, (int)W_ELEMS);

    const int gemmBlocks = (Mrows / 32) * (G3 / 64) / 8;   // 6144

    // ---- layer 0 ----
    gemm_bias_bf16<<<gemmBlocks, 256, 0, stream>>>(act, w0i, bih0, xg, Mrows, G3, In);
    zero_state<<<512, 256, 0, stream>>>(h, hbf, bar);
    gru_recurrent<<<32, 256, 0, stream>>>(xg, w0h, bhh0, h, hbf,
                                          /*y_bf=*/act, /*y_f=*/nullptr,
                                          h0_out, bar, Sn);
    // ---- layer 1 (consumes y0 in bf16 directly) ----
    gemm_bias_bf16<<<gemmBlocks, 256, 0, stream>>>(act, w1i, bih1, xg, Mrows, G3, Hn);
    zero_state<<<512, 256, 0, stream>>>(h, hbf, bar);
    gru_recurrent<<<32, 256, 0, stream>>>(xg, w1h, bhh1, h, hbf,
                                          /*y_bf=*/nullptr, /*y_f=*/y1,
                                          h1_out, bar, Sn);
}
